// QuantumInspiredLSTM_87041807221228
// MI455X (gfx1250) — matrix operand order repaired and checked by reading, not compiled, not run on hardware
//
#include <hip/hip_runtime.h>
#include <math.h>

// ---------------------------------------------------------------------------
// QuantumInspiredLSTM for MI455X (gfx1250, wave32, WMMA, async-to-LDS).
//   B=64, S=1024, I=64, H=256, L=2, O=1.
//   1) prep:    psi/Wcat pre-swizzled into WMMA B-fragment layout (f16)
//   2) quantum: Q = sqrt((x@cos th)^2 + (x@sin ph)^2)   [v_wmma_f32_16x16x32_f16]
//   3) scan x2: g = [x_t | h] @ Wcat ; gates live in accumulators, cell update
//               in registers; x_t double-buffered in LDS via
//               global_load_async_to_lds_b128 (+ s_wait_asynccnt)
//   4) fc:      out = h_last @ W_fc^T + b_fc
// (JAX threefry phase noise, sigma=0.01, is not reproducible on device; omitted.)
// ---------------------------------------------------------------------------

typedef __attribute__((ext_vector_type(16))) _Float16 v16h;
typedef __attribute__((ext_vector_type(8)))  _Float16 v8h;
typedef __attribute__((ext_vector_type(8)))  float    v8f;
typedef _Float16 f16;

#define BATCH 64
#define SEQ   1024
#define IDIM  64
#define HDIM  256
#define GDIM  1024   // 4*H
#define KCAT  512    // concat K: x-part (H) + h-part (H)

__device__ __forceinline__ v8f wmma16(v16h a, v16h b, v8f c) {
  return __builtin_amdgcn_wmma_f32_16x16x32_f16(
      false, a, false, b, (short)0, c, false, false);
}

__device__ __forceinline__ v16h cat8(v8h lo, v8h hi) {
  return __builtin_shufflevector(lo, hi, 0, 1, 2, 3, 4, 5, 6, 7,
                                         8, 9, 10, 11, 12, 13, 14, 15);
}

__device__ __forceinline__ float sigmoidf_(float x) {
  return 1.f / (1.f + __expf(-x));
}

// Async DMA: 16 bytes per lane, global -> LDS, tracked by ASYNCcnt.
__device__ __forceinline__ void async_b128_to_lds(void* lds_ptr, const void* gptr) {
  asm volatile("global_load_async_to_lds_b128 %0, %1, off"
               :: "v"((unsigned int)(uintptr_t)lds_ptr),
                  "v"((unsigned long long)(uintptr_t)gptr)
               : "memory");
}
__device__ __forceinline__ void wait_asynccnt0() {
  asm volatile("s_wait_asynccnt 0x0" ::: "memory");
}

// ---------------------------------------------------------------------------
// Kernel 1a: psi tables, written directly in WMMA B-fragment order:
//   frag index d = (((kt*16 + nt)*32 + lane)*16 + e)
//   maps to K = kt*32 + (lane>>4)*16 + e , N = nt*16 + (lane&15).
__global__ void prep_psi_kernel(const float* __restrict__ theta,
                                const float* __restrict__ phi,
                                f16* __restrict__ psiRSw, f16* __restrict__ psiISw) {
  int d = blockIdx.x * blockDim.x + threadIdx.x;
  if (d >= IDIM * HDIM) return;                 // 2*16*32*16 == 16384 == IDIM*HDIM
  int e  = d & 15;
  int L  = (d >> 4) & 31;
  int nt = (d >> 9) & 15;
  int kt = d >> 13;                             // 0..1
  int k  = kt * 32 + (L >> 4) * 16 + e;         // input dim
  int h  = nt * 16 + (L & 15);                  // hidden dim
  psiRSw[d] = (f16)cosf(theta[k * HDIM + h]);
  psiISw[d] = (f16)sinf(phi[k * HDIM + h]);
}

// Kernel 1b: Wcat = [W_ih ; W_hh]^T per layer, swizzled to B-fragment order:
//   d = ((((l*16 + kt)*64 + nt)*32 + lane)*16 + e)
__global__ void prep_w_kernel(const float* __restrict__ W_ih,
                              const float* __restrict__ W_hh,
                              f16* __restrict__ WSw) {
  int d = blockIdx.x * blockDim.x + threadIdx.x;
  if (d >= 2 * KCAT * GDIM) return;             // 2^20
  int e  = d & 15;
  int L  = (d >> 4) & 31;
  int nt = (d >> 9) & 63;
  int kt = (d >> 15) & 15;
  int l  = d >> 19;
  int k  = kt * 32 + (L >> 4) * 16 + e;         // 0..511
  int n  = nt * 16 + (L & 15);                  // 0..1023 (gate row)
  float v = (k < HDIM)
      ? W_ih[((size_t)l * GDIM + n) * HDIM + k]
      : W_hh[((size_t)l * GDIM + n) * HDIM + (k - HDIM)];
  WSw[d] = (f16)v;
}

__global__ void prep_bias_kernel(const float* __restrict__ b_ih,
                                 const float* __restrict__ b_hh,
                                 float* __restrict__ bsum) {
  int i = blockIdx.x * blockDim.x + threadIdx.x;
  if (i < 2 * GDIM) bsum[i] = b_ih[i] + b_hh[i];
}

// ---------------------------------------------------------------------------
// Kernel 2: quantum projection, one wave per 16x16 tile.
// M=B*S=65536, N=H=256, K=I=64 (2 WMMA k-steps); A loaded as 4x16B + convert.
__global__ void quantum_kernel(const float* __restrict__ x,      // (65536, 64)
                               const f16* __restrict__ psiRSw,
                               const f16* __restrict__ psiISw,
                               f16* __restrict__ Q) {             // (65536, 256)
  const int lane = threadIdx.x & 31;
  const int wid  = (blockIdx.x * blockDim.x + threadIdx.x) >> 5;
  const int NT   = HDIM / 16;
  const int m0   = (wid / NT) * 16;
  const int nt   = wid % NT;
  const int half = lane >> 4;
  const int mr   = lane & 15;
  const int nc   = lane & 15;

  const v16h* fragR = (const v16h*)psiRSw;
  const v16h* fragI = (const v16h*)psiISw;
  const float* xrow = x + (size_t)(m0 + mr) * IDIM;

  v8f accR = {}; v8f accI = {};
  #pragma unroll
  for (int kt = 0; kt < 2; ++kt) {
    const int k0 = kt * 32;
    float tmp[16];
    *(float4*)(tmp + 0)  = *(const float4*)(xrow + k0 + (half << 4));
    *(float4*)(tmp + 4)  = *(const float4*)(xrow + k0 + (half << 4) + 4);
    *(float4*)(tmp + 8)  = *(const float4*)(xrow + k0 + (half << 4) + 8);
    *(float4*)(tmp + 12) = *(const float4*)(xrow + k0 + (half << 4) + 12);
    v16h a;
    #pragma unroll
    for (int e = 0; e < 16; ++e) a[e] = (f16)tmp[e];
    v16h br = fragR[(kt * NT + nt) * 32 + lane];
    v16h bi = fragI[(kt * NT + nt) * 32 + lane];
    accR = wmma16(a, br, accR);
    accI = wmma16(a, bi, accI);
  }
  const int n0 = nt * 16;
  #pragma unroll
  for (int r = 0; r < 8; ++r) {
    int m = m0 + r + (half << 3);
    float yr = accR[r], yi = accI[r];
    Q[(size_t)m * HDIM + n0 + nc] = (f16)sqrtf(yr * yr + yi * yi);
  }
}

// ---------------------------------------------------------------------------
// Kernel 3: LSTM scan. One workgroup (1024 thr = 32 waves) loops over t.
// Wave w owns (m-pair = w>>4, n-group = w&15): two batch tiles x four gate
// tiles, so the full i/f/g/o quadruple for its 16 h-columns sits in its own
// accumulators -> cell update entirely in registers, h -> LDS f16.
// x_t is double-buffered in LDS via async-to-LDS DMA overlapped with WMMA.
// LDS: 2*32KB xbuf + 32KB h = 96 KB.
__global__ void lstm_scan_kernel(const f16* __restrict__ X,      // (B*S, H), row = b*SEQ + t
                                 const f16* __restrict__ WSw,    // swizzled (16,64,32,16)
                                 const float* __restrict__ bsum, // (1024)
                                 f16* __restrict__ Hout,         // (B*S, H)
                                 float* __restrict__ hlast,      // (64, 256) f32
                                 int store_all) {
  extern __shared__ char smem[];
  f16* xbuf0 = (f16*)smem;                 // 64*256 f16
  f16* xbuf1 = xbuf0 + BATCH * HDIM;
  f16* h_sh  = xbuf1 + BATCH * HDIM;       // 64*256 f16

  const int tid  = threadIdx.x;
  const int lane = tid & 31;
  const int w    = tid >> 5;
  const int half = lane >> 4;
  const int mr   = lane & 15;
  const int nc   = lane & 15;
  const int ng   = w & 15;                 // n-group: h-columns ng*16..+16
  const int mp   = w >> 4;                 // m-pair: batch tiles mp and mp+2
  const int ba   = mp * 16 + mr;           // A-row, tile a
  const int bb   = (mp + 2) * 16 + mr;     // A-row, tile b
  const int n    = ng * 16 + nc;           // owned h-column

  // Per-lane gate biases (constant over t).
  const float bi_n = bsum[n];
  const float bf_n = bsum[HDIM + n];
  const float bg_n = bsum[2 * HDIM + n];
  const float bo_n = bsum[3 * HDIM + n];

  float c0r[8], c1r[8];
  #pragma unroll
  for (int r = 0; r < 8; ++r) { c0r[r] = 0.f; c1r[r] = 0.f; }
  #pragma unroll
  for (int p = 0; p < 16; ++p) h_sh[tid + p * 1024] = (f16)0.f;

  // Staging: 64 rows x 512B = 2048 16B-chunks, 2 per thread.
  const int q0b = tid >> 5,          q0c = tid & 31;
  const int q1b = (tid + 1024) >> 5, q1c = tid & 31;

  // Prime buffer 0 with x_0.
  async_b128_to_lds(xbuf0 + q0b * HDIM + q0c * 8,
                    X + ((size_t)q0b * SEQ + 0) * HDIM + q0c * 8);
  async_b128_to_lds(xbuf0 + q1b * HDIM + q1c * 8,
                    X + ((size_t)q1b * SEQ + 0) * HDIM + q1c * 8);
  wait_asynccnt0();
  __syncthreads();

  const v16h* wfrag = (const v16h*)WSw;

  for (int t = 0; t < SEQ; ++t) {
    // Kick off async DMA of x_{t+1} into the other buffer (overlaps WMMA).
    if (t + 1 < SEQ) {
      f16* nb = ((t + 1) & 1) ? xbuf1 : xbuf0;
      async_b128_to_lds(nb + q0b * HDIM + q0c * 8,
                        X + ((size_t)q0b * SEQ + (t + 1)) * HDIM + q0c * 8);
      async_b128_to_lds(nb + q1b * HDIM + q1c * 8,
                        X + ((size_t)q1b * SEQ + (t + 1)) * HDIM + q1c * 8);
    }

    const f16* xb     = (t & 1) ? xbuf1 : xbuf0;
    const f16* rowa_x = xb + ba * HDIM;
    const f16* rowb_x = xb + bb * HDIM;
    const f16* rowa_h = h_sh + ba * HDIM - HDIM;   // absolute-K indexing
    const f16* rowb_h = h_sh + bb * HDIM - HDIM;

    v8f acc0[4] = {}; v8f acc1[4] = {};            // [gate] for tiles a, b
    #pragma unroll 1
    for (int kt = 0; kt < KCAT / 32; ++kt) {
      const int k0 = kt * 32;
      const f16* ra = (kt < 8) ? rowa_x : rowa_h;
      const f16* rb = (kt < 8) ? rowb_x : rowb_h;
      v16h a0 = cat8(*(const v8h*)(ra + k0 + (half << 4)),
                     *(const v8h*)(ra + k0 + 8 + (half << 4)));
      v16h a1 = cat8(*(const v8h*)(rb + k0 + (half << 4)),
                     *(const v8h*)(rb + k0 + 8 + (half << 4)));
      #pragma unroll
      for (int g = 0; g < 4; ++g) {
        v16h bv = wfrag[((size_t)kt * 64 + (g * 16 + ng)) * 32 + lane];
        acc0[g] = wmma16(a0, bv, acc0[g]);
        acc1[g] = wmma16(a1, bv, acc1[g]);
      }
    }

    // Cell update in registers (gate order i,f,g,o), h -> LDS + global.
    #pragma unroll
    for (int r = 0; r < 8; ++r) {
      int b = mp * 16 + r + (half << 3);
      float i_ = sigmoidf_(acc0[0][r] + bi_n);
      float f_ = sigmoidf_(acc0[1][r] + bf_n);
      float g_ = tanhf(acc0[2][r] + bg_n);
      float o_ = sigmoidf_(acc0[3][r] + bo_n);
      float c  = f_ * c0r[r] + i_ * g_;
      c0r[r] = c;
      float h = o_ * tanhf(c);
      h_sh[b * HDIM + n] = (f16)h;
      if (store_all) Hout[((size_t)b * SEQ + t) * HDIM + n] = (f16)h;
      if (t == SEQ - 1) hlast[b * HDIM + n] = h;
    }
    #pragma unroll
    for (int r = 0; r < 8; ++r) {
      int b = (mp + 2) * 16 + r + (half << 3);
      float i_ = sigmoidf_(acc1[0][r] + bi_n);
      float f_ = sigmoidf_(acc1[1][r] + bf_n);
      float g_ = tanhf(acc1[2][r] + bg_n);
      float o_ = sigmoidf_(acc1[3][r] + bo_n);
      float c  = f_ * c1r[r] + i_ * g_;
      c1r[r] = c;
      float h = o_ * tanhf(c);
      h_sh[b * HDIM + n] = (f16)h;
      if (store_all) Hout[((size_t)b * SEQ + t) * HDIM + n] = (f16)h;
      if (t == SEQ - 1) hlast[b * HDIM + n] = h;
    }

    if (t + 1 < SEQ) wait_asynccnt0();   // drain this wave's DMA
    __syncthreads();                     // publish h + staged x across waves
  }
}

// ---------------------------------------------------------------------------
// Kernel 4: final FC. out[b] = hlast[b,:] . W_fc[0,:] + b_fc[0]
__global__ void fc_kernel(const float* __restrict__ hlast,
                          const float* __restrict__ W_fc,
                          const float* __restrict__ b_fc,
                          float* __restrict__ out) {
  int b = threadIdx.x;
  if (b < BATCH) {
    float acc = b_fc[0];
    #pragma unroll 4
    for (int h = 0; h < HDIM; ++h) acc += hlast[b * HDIM + h] * W_fc[h];
    out[b] = acc;
  }
}

// ---------------------------------------------------------------------------
extern "C" void kernel_launch(void* const* d_in, const int* in_sizes, int n_in,
                              void* d_out, int out_size, void* d_ws, size_t ws_size,
                              hipStream_t stream) {
  const float* x     = (const float*)d_in[0];
  const float* theta = (const float*)d_in[1];
  const float* phi   = (const float*)d_in[2];
  const float* W_ih  = (const float*)d_in[3];
  const float* W_hh  = (const float*)d_in[4];
  const float* b_ih  = (const float*)d_in[5];
  const float* b_hh  = (const float*)d_in[6];
  const float* W_fc  = (const float*)d_in[7];
  const float* b_fc  = (const float*)d_in[8];
  float* out = (float*)d_out;

  char* ws = (char*)d_ws;
  size_t off = 0;
  auto carve = [&](size_t bytes) -> char* {
    char* p = ws + off;
    off = (off + bytes + 255) & ~(size_t)255;
    return p;
  };
  f16*   psiRSw = (f16*)carve(IDIM * HDIM * sizeof(f16));                 // 32 KB
  f16*   psiISw = (f16*)carve(IDIM * HDIM * sizeof(f16));                 // 32 KB
  f16*   WSw    = (f16*)carve((size_t)2 * KCAT * GDIM * sizeof(f16));     // 2 MB
  float* bsum   = (float*)carve((size_t)2 * GDIM * sizeof(float));        // 8 KB
  f16*   Q      = (f16*)carve((size_t)BATCH * SEQ * HDIM * sizeof(f16));  // 32 MB
  f16*   H1     = (f16*)carve((size_t)BATCH * SEQ * HDIM * sizeof(f16));  // 32 MB
  float* hlast  = (float*)carve((size_t)BATCH * HDIM * sizeof(float));    // 64 KB

  // 1) prep (fragment-swizzled weights)
  prep_psi_kernel<<<(IDIM * HDIM + 255) / 256, 256, 0, stream>>>(theta, phi, psiRSw, psiISw);
  prep_w_kernel<<<(2 * KCAT * GDIM + 255) / 256, 256, 0, stream>>>(W_ih, W_hh, WSw);
  prep_bias_kernel<<<(2 * GDIM + 255) / 256, 256, 0, stream>>>(b_ih, b_hh, bsum);

  // 2) quantum projection: 65536 wave-tiles, 8 waves per block.
  quantum_kernel<<<(BATCH * SEQ / 16) * (HDIM / 16) / 8, 256, 0, stream>>>(
      x, psiRSw, psiISw, Q);

  // 3) two LSTM scans, 96 KB LDS each.
  const size_t smem = (size_t)3 * BATCH * HDIM * sizeof(f16);
  lstm_scan_kernel<<<1, 1024, smem, stream>>>(Q,  WSw,               bsum,        H1, hlast, 1);
  lstm_scan_kernel<<<1, 1024, smem, stream>>>(H1, WSw + KCAT * GDIM, bsum + GDIM, H1, hlast, 0);

  // 4) head
  fc_kernel<<<1, 64, 0, stream>>>(hlast, W_fc, b_fc, out);
}
